// NLBlockND_cross_54709293416911
// MI455X (gfx1250) — matrix operand order repaired and checked by reading, not compiled, not run on hardware
//
#include <hip/hip_runtime.h>
#include <hip/hip_bf16.h>
#include <math.h>

#define SEQ 4096
#define BATCH 4
#define CH 64
#define ICH 32
#define BN_EPS 1e-5f

typedef __attribute__((ext_vector_type(16))) _Float16 v16h;
typedef __attribute__((ext_vector_type(8)))  _Float16 v8h;
typedef __attribute__((ext_vector_type(8)))  float    v8f;

// ---------------------------------------------------------------------------
// Kernel 1: 1x1-conv projections, fp32 -> f16, laid out for WMMA fragments.
//   Qf[b][n][i] = phi(x_other)   (A-matrix rows, row-major, 64B rows)
//   Kf[b][m][i] = theta(x_this)  (B-matrix "K^T" columns are contiguous rows)
//   Vt[b][i][m] = g(x_this)      (transposed so PV B-fragments are contiguous)
// ---------------------------------------------------------------------------
__global__ __launch_bounds__(256)
void nlb_proj_kernel(const float* __restrict__ x_this,
                     const float* __restrict__ x_other,
                     const float* __restrict__ g_w,  const float* __restrict__ g_b,
                     const float* __restrict__ t_w,  const float* __restrict__ t_b,
                     const float* __restrict__ p_w,  const float* __restrict__ p_b,
                     _Float16* __restrict__ Qf, _Float16* __restrict__ Kf,
                     _Float16* __restrict__ Vt)
{
    __shared__ float s_gw[ICH * CH], s_tw[ICH * CH], s_pw[ICH * CH];
    __shared__ float s_gb[ICH], s_tb[ICH], s_pb[ICH];
    for (int i = threadIdx.x; i < ICH * CH; i += blockDim.x) {
        s_gw[i] = g_w[i]; s_tw[i] = t_w[i]; s_pw[i] = p_w[i];
    }
    if (threadIdx.x < ICH) {
        s_gb[threadIdx.x] = g_b[threadIdx.x];
        s_tb[threadIdx.x] = t_b[threadIdx.x];
        s_pb[threadIdx.x] = p_b[threadIdx.x];
    }
    __syncthreads();

    const int idx = blockIdx.x * blockDim.x + threadIdx.x;   // 0 .. B*N-1
    const int b = idx >> 12;          // N == 4096
    const int n = idx & (SEQ - 1);

    float xv[CH];

    // ---- this-branch row -> theta (K rows) and g (V columns) ----
    {
        const float* xr = x_this + ((size_t)n * BATCH + b) * CH;
#pragma unroll
        for (int c = 0; c < CH; ++c) xv[c] = xr[c];
        for (int i = 0; i < ICH; ++i) {
            float t = s_tb[i], g = s_gb[i];
#pragma unroll
            for (int c = 0; c < CH; ++c) {
                t = fmaf(s_tw[i * CH + c], xv[c], t);
                g = fmaf(s_gw[i * CH + c], xv[c], g);
            }
            Kf[((size_t)(b * SEQ + n)) * ICH + i] = (_Float16)t;
            Vt[((size_t)(b * ICH + i)) * SEQ + n] = (_Float16)g;
        }
    }
    // ---- other-branch row -> phi (Q rows) ----
    {
        const float* xr = x_other + ((size_t)n * BATCH + b) * CH;
#pragma unroll
        for (int c = 0; c < CH; ++c) xv[c] = xr[c];
        for (int i = 0; i < ICH; ++i) {
            float p = s_pb[i];
#pragma unroll
            for (int c = 0; c < CH; ++c) p = fmaf(s_pw[i * CH + c], xv[c], p);
            Qf[((size_t)(b * SEQ + n)) * ICH + i] = (_Float16)p;
        }
    }
}

// Level-ordered butterfly max across the 16-lane half: issue 8 independent
// ds_swizzle ops per level so the DS pipe stays full (4 wait points total).
#define SWZ_MAX_LEVEL(rm, OFF)                                                  \
    {                                                                           \
        float t_[8];                                                            \
        _Pragma("unroll")                                                       \
        for (int r_ = 0; r_ < 8; ++r_)                                          \
            t_[r_] = __int_as_float(                                            \
                __builtin_amdgcn_ds_swizzle(__float_as_int(rm[r_]), OFF));      \
        _Pragma("unroll")                                                       \
        for (int r_ = 0; r_ < 8; ++r_) rm[r_] = fmaxf(rm[r_], t_[r_]);          \
    }

// ---------------------------------------------------------------------------
// Kernel 2: flash-attention. One wave owns a 16-row Q tile; iterates over
// 64-wide key/value blocks. 10 WMMAs per block: 4 score, 4 P.V, and 2 P.1
// (all-ones B matrix) which computes the softmax row-sum entirely on the
// matrix pipe -- the running "l" is just a third accumulator fragment whose
// value is replicated across all lanes, so no cross-lane sum is ever needed.
// ---------------------------------------------------------------------------
__global__ __launch_bounds__(128)
void nlb_attn_kernel(const _Float16* __restrict__ Qf,
                     const _Float16* __restrict__ Kf,
                     const _Float16* __restrict__ Vt,
                     float* __restrict__ Y)
{
    const int b     = blockIdx.y;
    const int wave  = threadIdx.x >> 5;
    const int lane  = threadIdx.x & 31;
    const int l15   = lane & 15;
    const int hi    = lane >> 4;              // which 16-lane half
    const int qbase = (blockIdx.x * 4 + wave) * 16;

    // per-wave P staging tile: 16 rows x 72 halves (144B stride, 16B aligned)
    __shared__ __align__(16) _Float16 ptile[4][16][72];

    // ----- load Q A-fragment (16x32 f16): lane l15 -> row, halves split by hi
    v16h qa;
    {
        const v8h* q8 = (const v8h*)(Qf + ((size_t)(b * SEQ + qbase + l15)) * ICH);
        v8h qlo = q8[2 * hi];        // K = hi*16 .. hi*16+7
        v8h qhi = q8[2 * hi + 1];    // K = hi*16+8 .. hi*16+15
        qa = __builtin_shufflevector(qlo, qhi, 0,1,2,3,4,5,6,7,8,9,10,11,12,13,14,15);
    }

    // all-ones B fragment: P x ones == row-sum broadcast to every column/lane
    v16h onesb;
#pragma unroll
    for (int k = 0; k < 16; ++k) onesb[k] = (_Float16)1.0f;

    float m8[8];
    v8f acc0 = {0.f,0.f,0.f,0.f,0.f,0.f,0.f,0.f};
    v8f acc1 = {0.f,0.f,0.f,0.f,0.f,0.f,0.f,0.f};
    v8f accl = {0.f,0.f,0.f,0.f,0.f,0.f,0.f,0.f};   // running row-sum l
    const v8f z = {0.f,0.f,0.f,0.f,0.f,0.f,0.f,0.f};
#pragma unroll
    for (int r = 0; r < 8; ++r) m8[r] = -3.0e38f;

    // per-lane fragment base pointers; all loads below use immediate offsets
    const _Float16* kp = Kf + ((size_t)(b * SEQ + l15)) * ICH + hi * 16;
    const _Float16* vp = Vt + ((size_t)(b * ICH + l15)) * SEQ + hi * 16;

    for (int it = 0; it < SEQ / 64; ++it) {
        // ---- K^T B-fragments, 4 groups of 16 columns (32B contiguous/lane)
        v16h kb[4];
#pragma unroll
        for (int j = 0; j < 4; ++j)
            kb[j] = *(const v16h*)(kp + j * (16 * ICH));
        // ---- V B-fragments: 2 position chunks x 2 IC halves
        v16h vb[4];
#pragma unroll
        for (int h = 0; h < 2; ++h)
#pragma unroll
            for (int j = 0; j < 2; ++j)
                vb[h * 2 + j] = *(const v16h*)(vp + j * (16 * SEQ) + 32 * h);
        kp += 64 * ICH;
        vp += 64;

        // ---- scores: 4 independent WMMAs
        v8f s[4];
#pragma unroll
        for (int j = 0; j < 4; ++j)
            s[j] = __builtin_amdgcn_wmma_f32_16x16x32_f16(false, qa, false, kb[j],
                                                          (short)0, z, false, false);

        // ---- block row-max: local 4-way combine, then batched butterfly
        float rm[8];
#pragma unroll
        for (int r = 0; r < 8; ++r)
            rm[r] = fmaxf(fmaxf(s[0][r], s[1][r]), fmaxf(s[2][r], s[3][r]));
        SWZ_MAX_LEVEL(rm, 0x041F)
        SWZ_MAX_LEVEL(rm, 0x081F)
        SWZ_MAX_LEVEL(rm, 0x101F)
        SWZ_MAX_LEVEL(rm, 0x201F)

        // ---- online rescale + exp (row sums handled by the P.1 WMMAs below)
#pragma unroll
        for (int r = 0; r < 8; ++r) {
            float mn = fmaxf(m8[r], rm[r]);
            float sc = __expf(m8[r] - mn);
            m8[r] = mn;
#pragma unroll
            for (int j = 0; j < 4; ++j) s[j][r] = __expf(s[j][r] - mn);
            acc0[r] *= sc;
            acc1[r] *= sc;
            accl[r] *= sc;
        }

        // ---- C-layout P -> LDS -> two A-layout fragments (16x32 each)
#pragma unroll
        for (int r = 0; r < 8; ++r) {
            const int row = hi * 8 + r;
#pragma unroll
            for (int j = 0; j < 4; ++j)
                ptile[wave][row][16 * j + l15] = (_Float16)s[j][r];
        }
        asm volatile("s_wait_dscnt 0" ::: "memory");   // wave-private RAW fence
        v16h pa0, pa1;
        {
            const v8h* pr = (const v8h*)&ptile[wave][l15][0];
            v8h a = pr[2 * hi],     c = pr[2 * hi + 1];   // columns cb .. cb+31
            v8h d = pr[4 + 2 * hi], e = pr[5 + 2 * hi];   // columns cb+32 .. cb+63
            pa0 = __builtin_shufflevector(a, c, 0,1,2,3,4,5,6,7,8,9,10,11,12,13,14,15);
            pa1 = __builtin_shufflevector(d, e, 0,1,2,3,4,5,6,7,8,9,10,11,12,13,14,15);
        }

        // ---- O += P.V (4 WMMAs) and l += P.1 (2 WMMAs)
        acc0 = __builtin_amdgcn_wmma_f32_16x16x32_f16(false, pa0, false, vb[0],
                                                      (short)0, acc0, false, false);
        acc1 = __builtin_amdgcn_wmma_f32_16x16x32_f16(false, pa0, false, vb[1],
                                                      (short)0, acc1, false, false);
        accl = __builtin_amdgcn_wmma_f32_16x16x32_f16(false, pa0, false, onesb,
                                                      (short)0, accl, false, false);
        acc0 = __builtin_amdgcn_wmma_f32_16x16x32_f16(false, pa1, false, vb[2],
                                                      (short)0, acc0, false, false);
        acc1 = __builtin_amdgcn_wmma_f32_16x16x32_f16(false, pa1, false, vb[3],
                                                      (short)0, acc1, false, false);
        accl = __builtin_amdgcn_wmma_f32_16x16x32_f16(false, pa1, false, onesb,
                                                      (short)0, accl, false, false);
    }

    // ---- normalize and write y[b][n][ic] (fp32); l replicated in every lane
#pragma unroll
    for (int r = 0; r < 8; ++r) {
        const int row = hi * 8 + r;
        const float inv = 1.0f / accl[r];
        float* yr = Y + ((size_t)(b * SEQ + qbase + row)) * ICH;
        yr[l15]      = acc0[r] * inv;
        yr[16 + l15] = acc1[r] * inv;
    }
}

// ---------------------------------------------------------------------------
// Kernel 3a: zero BN statistics buffer
// ---------------------------------------------------------------------------
__global__ void nlb_zero_kernel(float* __restrict__ p, int n)
{
    int i = blockIdx.x * blockDim.x + threadIdx.x;
    if (i < n) p[i] = 0.f;
}

// ---------------------------------------------------------------------------
// Kernel 3b: W_z 1x1 conv + per-channel partial BN sums (LDS reduce + atomics)
// ---------------------------------------------------------------------------
__global__ __launch_bounds__(256)
void nlb_wz_kernel(const float* __restrict__ Y,
                   const float* __restrict__ wz_w, const float* __restrict__ wz_b,
                   float* __restrict__ WY, float* __restrict__ stats)
{
    __shared__ float s_w[CH * ICH];
    __shared__ float s_b[CH];
    __shared__ float s_acc[2 * CH];
    for (int i = threadIdx.x; i < CH * ICH; i += 256) s_w[i] = wz_w[i];
    if (threadIdx.x < CH) s_b[threadIdx.x] = wz_b[threadIdx.x];
    if (threadIdx.x < 2 * CH) s_acc[threadIdx.x] = 0.f;
    __syncthreads();

    const int idx = blockIdx.x * 256 + threadIdx.x;   // == b*SEQ + n
    const int b = idx >> 12;
    const int n = idx & (SEQ - 1);

    float yv[ICH];
    const float* yr = Y + (size_t)idx * ICH;
#pragma unroll
    for (int i = 0; i < ICH; ++i) yv[i] = yr[i];

    for (int c = 0; c < CH; ++c) {
        float acc = s_b[c];
#pragma unroll
        for (int i = 0; i < ICH; ++i) acc = fmaf(s_w[c * ICH + i], yv[i], acc);
        WY[((size_t)(b * CH + c)) * SEQ + n] = acc;
        atomicAdd(&s_acc[c], acc);            // ds_add_f32
        atomicAdd(&s_acc[CH + c], acc * acc);
    }
    __syncthreads();
    if (threadIdx.x < CH) {
        atomicAdd(&stats[threadIdx.x],      s_acc[threadIdx.x]);
        atomicAdd(&stats[CH + threadIdx.x], s_acc[CH + threadIdx.x]);
    }
}

// ---------------------------------------------------------------------------
// Kernel 4: BatchNorm finalize + residual (x_this permuted to (B,C,N))
// ---------------------------------------------------------------------------
__global__ __launch_bounds__(256)
void nlb_bn_kernel(const float* __restrict__ WY, const float* __restrict__ stats,
                   const float* __restrict__ gamma, const float* __restrict__ beta,
                   const float* __restrict__ x_this, float* __restrict__ out)
{
    const size_t idx = (size_t)blockIdx.x * 256 + threadIdx.x;   // over B*C*N
    const int n = (int)(idx & (SEQ - 1));
    const int c = (int)((idx >> 12) & (CH - 1));
    const int b = (int)(idx >> 18);

    const float cnt  = (float)(BATCH * SEQ);
    const float mean = stats[c] / cnt;
    const float var  = stats[CH + c] / cnt - mean * mean;
    const float w    = WY[idx];
    const float res  = x_this[((size_t)n * BATCH + b) * CH + c];
    out[idx] = (w - mean) * rsqrtf(var + BN_EPS) * gamma[c] + beta[c] + res;
}

// ---------------------------------------------------------------------------
extern "C" void kernel_launch(void* const* d_in, const int* in_sizes, int n_in,
                              void* d_out, int out_size, void* d_ws, size_t ws_size,
                              hipStream_t stream) {
    (void)in_sizes; (void)n_in; (void)out_size; (void)ws_size;

    const float* x_this  = (const float*)d_in[0];
    const float* x_other = (const float*)d_in[1];
    const float* g_w     = (const float*)d_in[2];
    const float* g_b     = (const float*)d_in[3];
    const float* t_w     = (const float*)d_in[4];
    const float* t_b     = (const float*)d_in[5];
    const float* p_w     = (const float*)d_in[6];
    const float* p_b     = (const float*)d_in[7];
    const float* wz_w    = (const float*)d_in[8];
    const float* wz_b    = (const float*)d_in[9];
    const float* bn_g    = (const float*)d_in[10];
    const float* bn_b    = (const float*)d_in[11];
    float* out = (float*)d_out;

    char* ws = (char*)d_ws;
    const size_t halfmat = (size_t)BATCH * SEQ * ICH * sizeof(_Float16); // 1 MB
    _Float16* Qf = (_Float16*)(ws);
    _Float16* Kf = (_Float16*)(ws + halfmat);
    _Float16* Vt = (_Float16*)(ws + 2 * halfmat);
    float* Y     = (float*)(ws + 3 * halfmat);                            // 2 MB
    float* WY    = (float*)(ws + 3 * halfmat + (size_t)BATCH * SEQ * ICH * 4);
    float* stats = (float*)(ws + 3 * halfmat + (size_t)BATCH * SEQ * ICH * 4
                               + (size_t)BATCH * CH * SEQ * 4);

    // 1) projections -> f16 fragments
    nlb_proj_kernel<<<(BATCH * SEQ) / 256, 256, 0, stream>>>(
        x_this, x_other, g_w, g_b, t_w, t_b, p_w, p_b, Qf, Kf, Vt);

    // 2) fused flash attention (WMMA)
    nlb_attn_kernel<<<dim3(SEQ / 64, BATCH), 128, 0, stream>>>(Qf, Kf, Vt, Y);

    // 3) BN stats zero + W_z projection with partial sums
    nlb_zero_kernel<<<1, 128, 0, stream>>>(stats, 2 * CH);
    nlb_wz_kernel<<<(BATCH * SEQ) / 256, 256, 0, stream>>>(Y, wz_w, wz_b, WY, stats);

    // 4) BN finalize + residual
    nlb_bn_kernel<<<(BATCH * CH * SEQ) / 256, 256, 0, stream>>>(
        WY, stats, bn_g, bn_b, x_this, out);
}
